// DynamicReductionNetwork_25383256719895
// MI455X (gfx1250) — matrix operand order repaired and checked by reading, not compiled, not run on hardware
//
#include <hip/hip_runtime.h>
#include <stdint.h>

#define NB    32
#define NNODE 1024
#define HD    64
#define KNN   16
#define NODES (NB*NNODE)

typedef _Float16 h16;
typedef __attribute__((ext_vector_type(2)))  _Float16 h2v;
typedef __attribute__((ext_vector_type(16))) _Float16 v16h;
typedef __attribute__((ext_vector_type(8)))  float    v8f;

union FragA { v16h h; unsigned u[8]; };
union Pk2   { h2v v; unsigned u; };

__device__ __forceinline__ float eluf(float x){ return x > 0.f ? x : (__expf(x) - 1.f); }

__device__ __forceinline__ v8f wmma_f16(const FragA& a, const FragA& b, v8f c){
  return __builtin_amdgcn_wmma_f32_16x16x32_f16(false, a.h, false, b.h, (short)0, c, false, false);
}

// ---------------------------------------------------------------------------
// Stage 0: input MLP  x*norm -> 4->32->64->64 (ELU), emit f16 copy + |h|^2
// ---------------------------------------------------------------------------
__global__ __launch_bounds__(256) void k_input_mlp(
    const float* __restrict__ x, const float* __restrict__ norm,
    const float* __restrict__ W1, const float* __restrict__ B1,
    const float* __restrict__ W2, const float* __restrict__ B2,
    const float* __restrict__ W3, const float* __restrict__ B3,
    h16* __restrict__ h0h, float* __restrict__ x2o)
{
  __shared__ float sW1[4*32], sB1[32], sW2[32*64], sB2[64], sW3[64*64], sB3[64];
  int tid = threadIdx.x;
  for (int s = tid; s < 4*32;  s += 256) sW1[s] = W1[s];
  for (int s = tid; s < 32;    s += 256) sB1[s] = B1[s];
  for (int s = tid; s < 32*64; s += 256) sW2[s] = W2[s];
  for (int s = tid; s < 64;    s += 256) sB2[s] = B2[s];
  for (int s = tid; s < 64*64; s += 256) sW3[s] = W3[s];
  for (int s = tid; s < 64;    s += 256) sB3[s] = B3[s];
  __syncthreads();
  int node = blockIdx.x * 256 + tid;
  float xv[4];
  for (int q = 0; q < 4; ++q) xv[q] = x[node*4+q] * norm[q];
  float a1[32];
  for (int t = 0; t < 32; ++t) {
    float s = sB1[t];
    for (int q = 0; q < 4; ++q) s += xv[q] * sW1[q*32+t];
    a1[t] = eluf(s);
  }
  float a2[64];
  for (int t = 0; t < 64; ++t) {
    float s = sB2[t];
    for (int q = 0; q < 32; ++q) s += a1[q] * sW2[q*64+t];
    a2[t] = eluf(s);
  }
  float x2 = 0.f;
  for (int t = 0; t < 64; ++t) {
    float s = sB3[t];
    for (int q = 0; q < 64; ++q) s += a2[q] * sW3[q*64+t];
    s = eluf(s);
    h0h[(size_t)node*HD + t] = (h16)s;
    x2 += s*s;
  }
  x2o[node] = x2;
}

// ---------------------------------------------------------------------------
// kNN: WMMA Gram matrix + fused per-row top-16 selection.
// Block = 256 (8 waves), one block handles 16 rows of one graph.
// Wave w sweeps column tiles w, w+8, ...  Each tile: 2x v_wmma_f32_16x16x32_f16.
// Emits neighbor indices + directed adjacency bitmask (32 words/row).
// ---------------------------------------------------------------------------
__global__ __launch_bounds__(256) void k_knn(
    const h16* __restrict__ hh, const float* __restrict__ x2,
    const int* __restrict__ valid,
    int* __restrict__ idxOut, unsigned* __restrict__ kdir)
{
  __shared__ float topd[8][16][16];
  __shared__ int   topi[8][16][16];
  __shared__ float ctile[8][16][16];
  const float FINF = __builtin_inff();
  int tid = threadIdx.x, w = tid >> 5, lane = tid & 31;
  int b = blockIdx.x >> 6;                 // NNODE/16 == 64 tiles of rows
  int rowbase = (blockIdx.x & 63) * 16;
  for (int s = tid; s < 8*16*16; s += 256) {
    ((float*)topd)[s] = FINF;
    ((int*)topi)[s]   = 0x7fffffff;
  }
  __syncthreads();
  int m = lane & 15, half = lane >> 4;
  // A fragments: 16 rows x K=64 (two K-chunks of 32)
  const unsigned* rowp = reinterpret_cast<const unsigned*>(hh + (size_t)(b*NNODE + rowbase + m) * HD);
  FragA A0, A1;
  for (int i = 0; i < 8; ++i) {
    int k0 = half*16 + 2*i;
    A0.u[i] = rowp[k0 >> 1];
    A1.u[i] = rowp[(k0 + 32) >> 1];
  }
  float x2r[8];
  for (int r = 0; r < 8; ++r) x2r[r] = x2[b*NNODE + rowbase + r + 8*half];

  for (int t = w; t < 64; t += 8) {
    int colbase = t * 16;
    const unsigned* colp = reinterpret_cast<const unsigned*>(hh + (size_t)(b*NNODE + colbase + m) * HD);
    FragA B0, B1;
    for (int i = 0; i < 8; ++i) {
      int kb = half*16 + 2*i;
      B0.u[i] = colp[kb >> 1];
      B1.u[i] = colp[(kb + 32) >> 1];
    }
    v8f acc;
    for (int r = 0; r < 8; ++r) acc[r] = 0.f;
    acc = wmma_f16(A0, B0, acc);
    acc = wmma_f16(A1, B1, acc);
    float x2c = x2[b*NNODE + colbase + m];
    int   vc  = valid ? valid[b*NNODE + colbase + m] : 1;
    for (int r = 0; r < 8; ++r) {
      int mr = r + 8*half;
      float d = fmaxf(x2r[r] + x2c - 2.f*acc[r], 0.f);
      if ((rowbase + mr) == (colbase + m) || !vc) d = FINF;
      ctile[w][mr][m] = d;
    }
    __builtin_amdgcn_wave_barrier();
    if (lane < 16) {                       // per-row sorted insertion (ties -> lower idx)
      float* L  = topd[w][lane];
      int*   LI = topi[w][lane];
      for (int n = 0; n < 16; ++n) {
        float d = ctile[w][lane][n];
        int id = colbase + n;
        if (d < L[15] || (d == L[15] && id < LI[15])) {
          int p = 15;
          while (p > 0 && (d < L[p-1] || (d == L[p-1] && id < LI[p-1]))) {
            L[p] = L[p-1]; LI[p] = LI[p-1]; --p;
          }
          L[p] = d; LI[p] = id;
        }
      }
    }
    __builtin_amdgcn_wave_barrier();
  }
  __syncthreads();
  if (tid < 16) {                          // 8-way merge of sorted lists
    int row = tid;
    int hp[8] = {0,0,0,0,0,0,0,0};
    unsigned bits[32];
    for (int i = 0; i < 32; ++i) bits[i] = 0u;
    int gnode = b*NNODE + rowbase + row;
    for (int s = 0; s < KNN; ++s) {
      float bd = __builtin_inff(); int bi = 0x7fffffff; int bq = 0;
      for (int q = 0; q < 8; ++q) {
        float d = topd[q][row][hp[q]];
        int  id = topi[q][row][hp[q]];
        if (d < bd || (d == bd && id < bi)) { bd = d; bi = id; bq = q; }
      }
      hp[bq]++;
      idxOut[gnode*KNN + s] = bi;
      bits[bi >> 5] |= 1u << (bi & 31);
    }
    for (int i = 0; i < 32; ++i) kdir[(size_t)gnode*32 + i] = bits[i];
  }
}

// adj = kdir | kdir^T  (kdir copied in beforehand; this ORs the transpose bits)
__global__ __launch_bounds__(256) void k_adj_scatter(const int* __restrict__ idx, unsigned* __restrict__ adj)
{
  int node = blockIdx.x*256 + threadIdx.x;
  int b = node >> 10, i = node & 1023;
  for (int k = 0; k < KNN; ++k) {
    int j = idx[node*KNN + k];
    atomicOr(&adj[(size_t)(b*NNODE + j)*32 + (i >> 5)], 1u << (i & 31));
  }
}

__global__ __launch_bounds__(256) void k_invdeg(const unsigned* __restrict__ adj, float* __restrict__ invdeg)
{
  int node = blockIdx.x*256 + threadIdx.x;
  int c = 0;
  for (int w = 0; w < 32; ++w) c += __popc(adj[(size_t)node*32 + w]);
  invdeg[node] = c > 0 ? 1.f / (float)c : 0.f;
}

// ---------------------------------------------------------------------------
// EdgeConv: one wave owns one node's 16 edges (a 16-row tile).
// MLP 128->96 (ELU) ->64 (ELU) fully in WMMA; A fragments built on the fly
// from gathered f16 features; weights pre-transposed into LDS as f16.
// pass 0: forward messages [x_i, x_j-x_i], summed to node i.
// pass 1: reverse messages [x_j, x_i-x_j], scattered to j with w_rev mask.
// ---------------------------------------------------------------------------
__global__ __launch_bounds__(128) void k_edgeconv(
    const h16* __restrict__ hh, const int* __restrict__ idx,
    const unsigned* __restrict__ kdir, const int* __restrict__ valid,
    const float* __restrict__ W1, const float* __restrict__ B1,
    const float* __restrict__ W2, const float* __restrict__ B2,
    float* __restrict__ hOut)
{
  __shared__ h16   W1t[96*128];        // [n][k]   24.0 KB
  __shared__ h16   W2t[64*96];         // [n][k]   12.0 KB
  __shared__ float b1s[96];
  __shared__ float b2s[64];
  __shared__ float outbuf[4][16*64];   // per-wave scratch (mid f16 / out f32) 16 KB
  int tid = threadIdx.x;
  for (int s = tid; s < 96*128; s += 128) { int n = s >> 7, k = s & 127; W1t[s] = (h16)W1[k*96 + n]; }
  for (int s = tid; s < 64*96;  s += 128) { int n = s / 96, k = s % 96;  W2t[s] = (h16)W2[k*64 + n]; }
  for (int s = tid; s < 96; s += 128) b1s[s] = B1[s];
  for (int s = tid; s < 64; s += 128) b2s[s] = B2[s];
  __syncthreads();

  int w = tid >> 5, lane = tid & 31;
  int node = blockIdx.x*4 + w;
  int b = node >> 10, iloc = node & 1023;
  int m = lane & 15, half = lane >> 4;
  int j = idx[node*KNN + m];                         // this row's neighbor
  const h16* xi = hh + (size_t)node*HD;
  const h16* xj = hh + (size_t)(b*NNODE + j)*HD;
  const unsigned* xiu = reinterpret_cast<const unsigned*>(xi);
  const unsigned* xju = reinterpret_cast<const unsigned*>(xj);
  const h2v* xiv = reinterpret_cast<const h2v*>(xi);
  const h2v* xjv = reinterpret_cast<const h2v*>(xj);
  h16* mid = reinterpret_cast<h16*>(&outbuf[w][0]);  // 16 x 96 f16
  const unsigned* midu = reinterpret_cast<const unsigned*>(mid);
  const unsigned* W1u = reinterpret_cast<const unsigned*>(W1t);
  const unsigned* W2u = reinterpret_cast<const unsigned*>(W2t);
  float* outw = &outbuf[w][0];                       // 16 x 64 f32
  int vflag = valid ? valid[node] : 1;

  for (int pass = 0; pass < 2; ++pass) {
    FragA A[4];
    for (int c = 0; c < 4; ++c) {
      for (int i = 0; i < 8; ++i) {
        int k = c*32 + half*16 + 2*i;
        unsigned u;
        if (k < 64) {
          u = (pass == 0) ? xiu[k >> 1] : xju[k >> 1];
        } else {
          int q = (k - 64) >> 1;
          Pk2 pk;
          pk.v = (pass == 0) ? (xjv[q] - xiv[q]) : (xiv[q] - xjv[q]);
          u = pk.u;
        }
        A[c].u[i] = u;
      }
    }
    // layer 1: 16x128 @ 128x96
    for (int t = 0; t < 6; ++t) {
      int n = t*16 + m;
      v8f acc;
      float bv = b1s[n];
      for (int r = 0; r < 8; ++r) acc[r] = bv;
      for (int c = 0; c < 4; ++c) {
        FragA Bf;
        for (int i = 0; i < 8; ++i) {
          int kk = c*32 + half*16 + 2*i;
          Bf.u[i] = W1u[(n*128 + kk) >> 1];
        }
        acc = wmma_f16(A[c], Bf, acc);
      }
      for (int r = 0; r < 8; ++r)
        mid[(r + 8*half)*96 + n] = (h16)eluf(acc[r]);
    }
    __syncthreads();
    // layer 2: 16x96 @ 96x64
    FragA A2[3];
    for (int c = 0; c < 3; ++c)
      for (int i = 0; i < 8; ++i) {
        int k = c*32 + half*16 + 2*i;
        A2[c].u[i] = midu[(m*96 + k) >> 1];
      }
    v8f acc2[4];
    for (int t = 0; t < 4; ++t) {
      int n = t*16 + m;
      float bv = b2s[n];
      for (int r = 0; r < 8; ++r) acc2[t][r] = bv;
      for (int c = 0; c < 3; ++c) {
        FragA Bf;
        for (int i = 0; i < 8; ++i) {
          int kk = c*32 + half*16 + 2*i;
          Bf.u[i] = W2u[(n*96 + kk) >> 1];
        }
        acc2[t] = wmma_f16(A2[c], Bf, acc2[t]);
      }
    }
    __syncthreads();                                 // mid fully consumed
    for (int t = 0; t < 4; ++t) {
      int n = t*16 + m;
      for (int r = 0; r < 8; ++r)
        outw[(r + 8*half)*64 + n] = eluf(acc2[t][r]);
    }
    __syncthreads();
    if (pass == 0) {                                 // sum over the 16 fwd messages
      int c0 = lane*2;
      float s0 = 0.f, s1 = 0.f;
      for (int r = 0; r < 16; ++r) { s0 += outw[r*64 + c0]; s1 += outw[r*64 + c0 + 1]; }
      atomicAdd(&hOut[(size_t)node*HD + c0],     s0);
      atomicAdd(&hOut[(size_t)node*HD + c0 + 1], s1);
    } else {                                         // scatter reverse messages
      if (lane < 16) {
        unsigned word = kdir[(size_t)(b*NNODE + j)*32 + (iloc >> 5)];
        int mutual = (word >> (iloc & 31)) & 1;
        if (vflag && !mutual) {
          float* dst = hOut + (size_t)(b*NNODE + j)*HD;
          for (int c = 0; c < HD; ++c) atomicAdd(&dst[c], outw[lane*64 + c]);
        }
      }
    }
    __syncthreads();
  }
}

// ---------------------------------------------------------------------------
// graclus: greedy heavy-edge matching, one 1024-thread block per graph.
// Argmax tree with first-index tie-break to match jnp.argmax.
// ---------------------------------------------------------------------------
__global__ __launch_bounds__(1024) void k_graclus(
    const float* __restrict__ h1, const unsigned* __restrict__ adj,
    const float* __restrict__ invdeg,
    int* __restrict__ clusterOut, int* __restrict__ partnerOut)
{
  __shared__ int   unm[1024];
  __shared__ int   clu[1024];
  __shared__ int   par[1024];
  __shared__ float redv[1024];
  __shared__ int   redi[1024];
  __shared__ float xi[64];
  const float FINF = __builtin_inff();
  int b = blockIdx.x, j = threadIdx.x;
  unm[j] = 1; clu[j] = j; par[j] = j;
  __syncthreads();
  for (int i = 0; i < NNODE; ++i) {
    if (j < 64) xi[j] = h1[(size_t)(b*NNODE + i)*HD + j];
    __syncthreads();
    float wv = -FINF;
    if (unm[i] && unm[j] && j != i) {
      unsigned word = adj[(size_t)(b*NNODE + i)*32 + (j >> 5)];
      if ((word >> (j & 31)) & 1u) {
        const float* xjp = h1 + (size_t)(b*NNODE + j)*HD;
        float s = 0.f;
        for (int c = 0; c < HD; ++c) { float d = xi[c] - xjp[c]; s += d*d; }
        wv = sqrtf(s) * (invdeg[b*NNODE + i] + invdeg[b*NNODE + j]);
      }
    }
    redv[j] = wv; redi[j] = j;
    __syncthreads();
    for (int s = 512; s > 0; s >>= 1) {
      if (j < s) {
        float av = redv[j], bv = redv[j+s];
        int ai = redi[j], bi = redi[j+s];
        if (bv > av || (bv == av && bi < ai)) { redv[j] = bv; redi[j] = bi; }
      }
      __syncthreads();
    }
    if (j == 0) {
      int jj = redi[0]; float best = redv[0];
      if (unm[i] && best > -FINF) {
        int r = i < jj ? i : jj;
        clu[i] = r; clu[jj] = r;
        par[i] = jj; par[jj] = i;
        unm[jj] = 0;
      }
      unm[i] = 0;
    }
    __syncthreads();
  }
  clusterOut[b*NNODE + j] = clu[j];
  partnerOut[b*NNODE + j] = par[j];
}

// max-pool over matched pairs (each cluster has <= 2 members, rep = min index)
__global__ __launch_bounds__(256) void k_pool(
    const float* __restrict__ h1, const int* __restrict__ cluster, const int* __restrict__ partner,
    h16* __restrict__ h1ph, float* __restrict__ h1p,
    float* __restrict__ x2o, int* __restrict__ valid1)
{
  int node = blockIdx.x*256 + threadIdx.x;
  int b = node >> 10, i = node & 1023;
  if (cluster[node] == i) {
    int p = partner[node];
    const float* s1 = h1 + (size_t)node*HD;
    const float* s2 = h1 + (size_t)(b*NNODE + p)*HD;
    float x2 = 0.f;
    for (int t = 0; t < HD; ++t) {
      float v = fmaxf(s1[t], s2[t]);
      h1p[(size_t)node*HD + t]  = v;
      h1ph[(size_t)node*HD + t] = (h16)v;
      x2 += v*v;
    }
    x2o[node] = x2;
    valid1[node] = 1;
  } else {
    for (int t = 0; t < HD; ++t) {
      h1p[(size_t)node*HD + t]  = 0.f;
      h1ph[(size_t)node*HD + t] = (h16)0.f;
    }
    x2o[node] = 0.f;
    valid1[node] = 0;
  }
}

// global max over valid nodes + output MLP 64->64->32->2
__global__ __launch_bounds__(64) void k_final(
    const float* __restrict__ h2, const int* __restrict__ valid1,
    const float* __restrict__ W1, const float* __restrict__ B1,
    const float* __restrict__ W2, const float* __restrict__ B2,
    const float* __restrict__ W3, const float* __restrict__ B3,
    float* __restrict__ out)
{
  __shared__ float g[64];
  __shared__ float a1[64];
  __shared__ float a2[32];
  int b = blockIdx.x, t = threadIdx.x;
  float mx = -__builtin_inff();
  for (int n = 0; n < NNODE; ++n)
    if (valid1[b*NNODE + n]) mx = fmaxf(mx, h2[(size_t)(b*NNODE + n)*HD + t]);
  g[t] = mx;
  __syncthreads();
  float s = B1[t];
  for (int q = 0; q < 64; ++q) s += g[q] * W1[q*64 + t];
  a1[t] = eluf(s);
  __syncthreads();
  if (t < 32) {
    float s2 = B2[t];
    for (int q = 0; q < 64; ++q) s2 += a1[q] * W2[q*32 + t];
    a2[t] = eluf(s2);
  }
  __syncthreads();
  if (t < 2) {
    float s3 = B3[t];
    for (int q = 0; q < 32; ++q) s3 += a2[q] * W3[q*2 + t];
    out[b*2 + t] = s3;
  }
}

// ---------------------------------------------------------------------------
extern "C" void kernel_launch(void* const* d_in, const int* in_sizes, int n_in,
                              void* d_out, int out_size, void* d_ws, size_t ws_size,
                              hipStream_t stream)
{
  (void)in_sizes; (void)n_in; (void)out_size; (void)ws_size;
  const float* x     = (const float*)d_in[0];
  const float* norm  = (const float*)d_in[1];
  const float* in1_W = (const float*)d_in[2];
  const float* in1_b = (const float*)d_in[3];
  const float* in2_W = (const float*)d_in[4];
  const float* in2_b = (const float*)d_in[5];
  const float* in3_W = (const float*)d_in[6];
  const float* in3_b = (const float*)d_in[7];
  const float* c1a_W = (const float*)d_in[8];
  const float* c1a_b = (const float*)d_in[9];
  const float* c1b_W = (const float*)d_in[10];
  const float* c1b_b = (const float*)d_in[11];
  const float* c2a_W = (const float*)d_in[12];
  const float* c2a_b = (const float*)d_in[13];
  const float* c2b_W = (const float*)d_in[14];
  const float* c2b_b = (const float*)d_in[15];
  const float* o1_W  = (const float*)d_in[16];
  const float* o1_b  = (const float*)d_in[17];
  const float* o2_W  = (const float*)d_in[18];
  const float* o2_b  = (const float*)d_in[19];
  const float* o3_W  = (const float*)d_in[20];
  const float* o3_b  = (const float*)d_in[21];

  char* wsb = (char*)d_ws;
  size_t off = 0;
  auto carve = [&](size_t bytes) -> void* {
    void* p = wsb + off;
    off = (off + bytes + 255) & ~(size_t)255;
    return p;
  };
  h16*      h0h    = (h16*)     carve((size_t)NODES*HD*2);
  float*    x2a    = (float*)   carve((size_t)NODES*4);
  int*      idx1   = (int*)     carve((size_t)NODES*KNN*4);
  unsigned* k1dir  = (unsigned*)carve((size_t)NODES*32*4);
  unsigned* adj1   = (unsigned*)carve((size_t)NODES*32*4);
  float*    invdeg = (float*)   carve((size_t)NODES*4);
  float*    h1     = (float*)   carve((size_t)NODES*HD*4);
  int*      clus   = (int*)     carve((size_t)NODES*4);
  int*      part   = (int*)     carve((size_t)NODES*4);
  float*    h1p    = (float*)   carve((size_t)NODES*HD*4);
  h16*      h1ph   = (h16*)     carve((size_t)NODES*HD*2);
  float*    x2b    = (float*)   carve((size_t)NODES*4);
  int*      valid1 = (int*)     carve((size_t)NODES*4);
  int*      idx2   = (int*)     carve((size_t)NODES*KNN*4);
  unsigned* k2dir  = (unsigned*)carve((size_t)NODES*32*4);
  float*    h2b    = (float*)   carve((size_t)NODES*HD*4);

  k_input_mlp<<<NODES/256, 256, 0, stream>>>(x, norm, in1_W, in1_b, in2_W, in2_b, in3_W, in3_b, h0h, x2a);
  k_knn<<<NB*(NNODE/16), 256, 0, stream>>>(h0h, x2a, nullptr, idx1, k1dir);
  hipMemcpyAsync(adj1, k1dir, (size_t)NODES*32*4, hipMemcpyDeviceToDevice, stream);
  k_adj_scatter<<<NODES/256, 256, 0, stream>>>(idx1, adj1);
  k_invdeg<<<NODES/256, 256, 0, stream>>>(adj1, invdeg);
  hipMemsetAsync(h1, 0, (size_t)NODES*HD*4, stream);
  k_edgeconv<<<NODES/4, 128, 0, stream>>>(h0h, idx1, k1dir, nullptr, c1a_W, c1a_b, c1b_W, c1b_b, h1);
  k_graclus<<<NB, 1024, 0, stream>>>(h1, adj1, invdeg, clus, part);
  k_pool<<<NODES/256, 256, 0, stream>>>(h1, clus, part, h1ph, h1p, x2b, valid1);
  k_knn<<<NB*(NNODE/16), 256, 0, stream>>>(h1ph, x2b, valid1, idx2, k2dir);
  hipMemsetAsync(h2b, 0, (size_t)NODES*HD*4, stream);
  k_edgeconv<<<NODES/4, 128, 0, stream>>>(h1ph, idx2, k2dir, valid1, c2a_W, c2a_b, c2b_W, c2b_b, h2b);
  k_final<<<NB, 64, 0, stream>>>(h2b, valid1, o1_W, o1_b, o2_W, o2_b, o3_W, o3_b, (float*)d_out);
}
